// arneodobouros_66168266162518
// MI455X (gfx1250) — compile-verified, matrix operand order checked by reading, not run on hardware
//
#include <hip/hip_runtime.h>
#include <cstddef>

typedef __attribute__((ext_vector_type(16))) _Float16 v16h;
typedef __attribute__((ext_vector_type(8)))  float    v8f;

#define BSZ   8
#define LSEQ  16384
#define LP    16376            // L - 8
#define TOKS  (BSZ * LP)       // 131008 == 16 * 8188 (exactly tiled!)
#define TILES (TOKS / 16)      // 8188 full tiles, no remainder
#define ED    4
#define NST   16
#define DM    4

struct Layer {
  const float* in_proj;   // (8,4)
  const float* conv_w;    // (4,4)
  const float* conv_b;    // (4,)
  const float* x_proj;    // (33,4)
  const float* dt_w;      // (4,1)
  const float* dt_b;      // (4,)
  const float* A_log;     // (4,16)
  const float* Dp;        // (4,)
  const float* out_proj;  // (4,4)
  const float* norm_w;    // (4,)
};

__device__ __forceinline__ float siluf(float v)     { return v / (1.f + __expf(-v)); }
__device__ __forceinline__ float softplusf(float v) { return (v > 20.f) ? v : log1pf(__expf(v)); }

// ---------------- K0: 9-tap derivative, z and x_in = [z, flip(z)] ----------------
__global__ void k_prep(const float* __restrict__ x, float* __restrict__ z,
                       float* __restrict__ xin) {
  int t = blockIdx.x * blockDim.x + threadIdx.x;
  if (t >= TOKS) return;
  int b = t / LP, l = t % LP;
  const float* xb = x + (size_t)b * LSEQ;
  const float C0 = 1.f/280.f, C1 = -4.f/105.f, C2 = 0.2f, C3 = -0.8f;
  const float C5 = 0.8f, C6 = -0.2f, C7 = 4.f/105.f, C8 = -1.f/280.f;
  float d1 = C0*xb[l]   + C1*xb[l+1] + C2*xb[l+2] + C3*xb[l+3]
           + C5*xb[l+5] + C6*xb[l+6] + C7*xb[l+7] + C8*xb[l+8];
  float z1 = xb[l + 4];
  int lf = LP - 1 - l;
  float d2 = C0*xb[lf]   + C1*xb[lf+1] + C2*xb[lf+2] + C3*xb[lf+3]
           + C5*xb[lf+5] + C6*xb[lf+6] + C7*xb[lf+7] + C8*xb[lf+8];
  float z1f = xb[lf + 4];
  z[(size_t)t*2 + 0] = z1;
  z[(size_t)t*2 + 1] = d1;
  xin[(size_t)t*4 + 0] = z1;
  xin[(size_t)t*4 + 1] = d1;
  xin[(size_t)t*4 + 2] = z1f;
  xin[(size_t)t*4 + 3] = d2;
}

// ---------------- K1: rmsnorm + in_proj via WMMA (16 tokens / wave) --------------
// buf1[t*8 + 0..3] = xc0 (pre-conv), buf1[t*8 + 4..7] = zg
__global__ void k_proj_in(const float* __restrict__ u, const Layer lp,
                          float* __restrict__ buf1) {
  int tile = blockIdx.x;
  int tid  = threadIdx.x;          // 0..31
  int m    = tid & 15;             // token row within tile
  int t    = tile * 16 + m;        // always < TOKS (exact tiling)

  float v0 = u[(size_t)t*4+0], v1 = u[(size_t)t*4+1];
  float v2 = u[(size_t)t*4+2], v3 = u[(size_t)t*4+3];
  float ms = (v0*v0 + v1*v1 + v2*v2 + v3*v3) * 0.25f;
  float r  = rsqrtf(ms + 1e-5f);
  v0 = v0 * r * lp.norm_w[0];  v1 = v1 * r * lp.norm_w[1];
  v2 = v2 * r * lp.norm_w[2];  v3 = v3 * r * lp.norm_w[3];

  v16h a; v16h bb;
  #pragma unroll
  for (int i = 0; i < 16; ++i) { a[i] = (_Float16)0.f; bb[i] = (_Float16)0.f; }
  if (tid < 16) {                       // lanes 0-15 hold K=0..3 (rest of K zero-padded)
    a[0] = (_Float16)v0; a[1] = (_Float16)v1; a[2] = (_Float16)v2; a[3] = (_Float16)v3;
    if (tid < 8) {                      // B column n = tid, rows K=0..3 = in_proj[n][k]
      #pragma unroll
      for (int k = 0; k < 4; ++k) bb[k] = (_Float16)lp.in_proj[tid*4 + k];
    }
  }
  v8f c;
  #pragma unroll
  for (int i = 0; i < 8; ++i) c[i] = 0.f;
  v8f d = __builtin_amdgcn_wmma_f32_16x16x32_f16(false, a, false, bb,
                                                 (short)0, c, false, false);
  int f     = tid & 15;                 // output feature (column)
  int mbase = (tid < 16) ? 0 : 8;       // rows 0-7 vs 8-15
  if (f < 8) {
    size_t base = (size_t)(tile * 16 + mbase) * 8 + f;
    #pragma unroll
    for (int rr = 0; rr < 8; ++rr) buf1[base + (size_t)rr * 8] = d[rr];
  }
}

// ---------------- K2: causal conv + silu + x_proj via 3× WMMA --------------------
// buf2[t*40 + 0..3] = xc (post conv+silu); buf2[t*40 + 4..36] = dbc[0..32]
__global__ void k_conv_xproj(const float* __restrict__ buf1, const Layer lp,
                             float* __restrict__ buf2) {
  int tile = blockIdx.x;
  int tid  = threadIdx.x;
  int m    = tid & 15;
  int t    = tile * 16 + m;             // always < TOKS
  int b = t / LP, l = t % LP;

  float xc[4];
  #pragma unroll
  for (int e = 0; e < 4; ++e) {
    float s = lp.conv_b[e];
    #pragma unroll
    for (int k = 0; k < 4; ++k) {
      int li = l - 3 + k;
      if (li >= 0) s += buf1[((size_t)b*LP + li)*8 + e] * lp.conv_w[e*4 + k];
    }
    xc[e] = siluf(s);
  }
  if (tid < 16) {
    #pragma unroll
    for (int e = 0; e < 4; ++e) buf2[(size_t)t*40 + e] = xc[e];
  }

  v16h a;
  #pragma unroll
  for (int i = 0; i < 16; ++i) a[i] = (_Float16)0.f;
  if (tid < 16) {
    a[0] = (_Float16)xc[0]; a[1] = (_Float16)xc[1];
    a[2] = (_Float16)xc[2]; a[3] = (_Float16)xc[3];
  }
  v8f c;
  #pragma unroll
  for (int i = 0; i < 8; ++i) c[i] = 0.f;

  int mbase = (tid < 16) ? 0 : 8;
  #pragma unroll
  for (int g = 0; g < 3; ++g) {         // N-tiles over 33 output features
    v16h bb;
    #pragma unroll
    for (int i = 0; i < 16; ++i) bb[i] = (_Float16)0.f;
    int fi = g * 16 + (tid & 15);
    if (tid < 16 && fi < 33) {
      #pragma unroll
      for (int k = 0; k < 4; ++k) bb[k] = (_Float16)lp.x_proj[fi*4 + k];
    }
    v8f d = __builtin_amdgcn_wmma_f32_16x16x32_f16(false, a, false, bb,
                                                   (short)0, c, false, false);
    int fo = g * 16 + (tid & 15);
    if (fo < 33) {
      size_t base = (size_t)(tile * 16 + mbase) * 40 + 4 + fo;
      #pragma unroll
      for (int rr = 0; rr < 8; ++rr) buf2[base + (size_t)rr * 40] = d[rr];
    }
  }
}

// ---------------- K3: selective-scan, one wave per batch -------------------------
__global__ void k_scan(const float* __restrict__ u_in, const float* __restrict__ buf1,
                       const float* __restrict__ buf2, const Layer lp,
                       float* __restrict__ u_out) {
  int b   = blockIdx.x;
  int tid = threadIdx.x;     // 0..31
  int n   = tid & 15;
  int eh  = tid >> 4;        // lane handles states (eh, n) and (eh+2, n)
  int e0 = eh, e1 = eh + 2;

  float A0 = -__expf(lp.A_log[e0*16 + n]);
  float A1 = -__expf(lp.A_log[e1*16 + n]);
  float dtw0 = lp.dt_w[e0], dtb0 = lp.dt_b[e0];
  float dtw1 = lp.dt_w[e1], dtb1 = lp.dt_b[e1];
  float Dv[4];  float opv[16];
  #pragma unroll
  for (int i = 0; i < 4; ++i)  Dv[i]  = lp.Dp[i];
  #pragma unroll
  for (int i = 0; i < 16; ++i) opv[i] = lp.out_proj[i];

  float h0 = 0.f, h1 = 0.f;
  for (int l = 0; l < LP; ++l) {
    size_t t = (size_t)b * LP + l;
    const float* rec = buf2 + t * 40;
    __builtin_prefetch(rec + 40 * 8, 0, 3);           // global_prefetch ahead of chain
    __builtin_prefetch(buf1 + (t + 8) * 8, 0, 3);

    float4 xc4 = *(const float4*)rec;                 // 16B-aligned (stride 160B)
    float xcv[4] = { xc4.x, xc4.y, xc4.z, xc4.w };
    float dbc0 = rec[4];
    float Bm = rec[5 + n], Cm = rec[21 + n];

    float delta0 = softplusf(dbc0 * dtw0 + dtb0);
    float delta1 = softplusf(dbc0 * dtw1 + dtb1);
    h0 = __expf(delta0 * A0) * h0 + delta0 * Bm * xcv[e0];
    h1 = __expf(delta1 * A1) * h1 + delta1 * Bm * xcv[e1];

    float s0 = h0 * Cm, s1 = h1 * Cm;
    #pragma unroll
    for (int off = 8; off >= 1; off >>= 1) {          // reduce over N within 16 lanes
      s0 += __shfl_xor(s0, off, 32);
      s1 += __shfl_xor(s1, off, 32);
    }
    float o0 = __shfl_xor(s0, 16, 32);
    float o1 = __shfl_xor(s1, 16, 32);
    float y[4];
    if (eh == 0) { y[0] = s0; y[2] = s1; y[1] = o0; y[3] = o1; }
    else         { y[1] = s0; y[3] = s1; y[0] = o0; y[2] = o1; }

    float4 zg4 = *(const float4*)(buf1 + t * 8 + 4);  // 16B-aligned
    float zgv[4] = { zg4.x, zg4.y, zg4.z, zg4.w };
    float outv[4] = {0.f, 0.f, 0.f, 0.f};
    #pragma unroll
    for (int e = 0; e < 4; ++e) {
      float ye = y[e] + Dv[e] * xcv[e];
      float ge = ye * siluf(zgv[e]);
      #pragma unroll
      for (int dch = 0; dch < 4; ++dch) outv[dch] += ge * opv[dch*4 + e];
    }
    if (tid < 4) u_out[t*4 + tid] = u_in[t*4 + tid] + outv[tid];
  }
}

// ---------------- K4: controls = [aC, smooth(|bC|,10), smooth(|dC|,10)] ----------
__global__ void k_ctrl(const float* __restrict__ aC, const float* __restrict__ bC,
                       const float* __restrict__ dC, float* __restrict__ dout) {
  int i = blockIdx.x * blockDim.x + threadIdx.x;
  if (i >= TOKS * 4) return;
  int c = i & 3, t = i >> 2;
  int b = t / LP, l = t % LP;
  float* ctrl = dout + TOKS;
  ctrl[i] = aC[i];
  float sb = 0.f, sd = 0.f;
  #pragma unroll
  for (int j = -5; j <= 4; ++j) {                     // window 10, edge-clamped
    int lj = l + j;
    lj = lj < 0 ? 0 : (lj >= LP ? LP - 1 : lj);
    size_t idx = ((size_t)b * LP + lj) * 4 + c;
    sb += fabsf(bC[idx]);
    sd += fabsf(dC[idx]);
  }
  ctrl[(size_t)TOKS*4 + i] = sb * 0.1f;
  ctrl[(size_t)TOKS*8 + i] = sd * 0.1f;
}

// ---------------- K5: heads + yhat + trailing zero -------------------------------
__global__ void k_head(const float* __restrict__ aC, const float* __restrict__ z,
                       const float* __restrict__ dtp,
                       const float* __restrict__ anw, const float* __restrict__ anb,
                       const float* __restrict__ bnw, const float* __restrict__ bnb,
                       const float* __restrict__ dnw, const float* __restrict__ dnb,
                       const float* __restrict__ taup, float* __restrict__ dout) {
  int t = blockIdx.x * blockDim.x + threadIdx.x;
  if (t >= TOKS) return;
  const float* ctrl = dout + TOKS;
  float alpha = anb[0], beta = bnb[0], dd = dnb[0];
  #pragma unroll
  for (int c = 0; c < 4; ++c) {
    alpha += aC[(size_t)t*4 + c] * anw[c];
    beta  += ctrl[(size_t)TOKS*4 + (size_t)t*4 + c] * bnw[c];
    dd    += ctrl[(size_t)TOKS*8 + (size_t)t*4 + c] * dnw[c];
  }
  dd = fmaxf(dd, 0.f);
  float dtv = dtp[0], tau = taup[0];
  float z1 = z[(size_t)t*2], z2 = z[(size_t)t*2 + 1] / dtv;
  float yhat = alpha + beta*beta*z1 + z1*z1 - z1*z1*z1
             - z1*z2/tau - z1*z1*z2/tau - dd;
  dout[t] = yhat;
  if (t == 0) dout[(size_t)TOKS * 13] = 0.f;          // third output: zeros((1,))
}

extern "C" void kernel_launch(void* const* d_in, const int* in_sizes, int n_in,
                              void* d_out, int out_size, void* d_ws, size_t ws_size,
                              hipStream_t stream) {
  (void)in_sizes; (void)n_in; (void)out_size; (void)ws_size;
  const float* x   = (const float*)d_in[0];
  const float* dtp = (const float*)d_in[1];

  Layer layers[6];
  int idx = 2;
  for (int i = 0; i < 6; ++i) {       // alpha(2), beta(2), d(2); 10 arrays each
    layers[i].in_proj  = (const float*)d_in[idx++];
    layers[i].conv_w   = (const float*)d_in[idx++];
    layers[i].conv_b   = (const float*)d_in[idx++];
    layers[i].x_proj   = (const float*)d_in[idx++];
    layers[i].dt_w     = (const float*)d_in[idx++];
    layers[i].dt_b     = (const float*)d_in[idx++];
    layers[i].A_log    = (const float*)d_in[idx++];
    layers[i].Dp       = (const float*)d_in[idx++];
    layers[i].out_proj = (const float*)d_in[idx++];
    layers[i].norm_w   = (const float*)d_in[idx++];
  }
  const float* anw  = (const float*)d_in[idx++];
  const float* anb  = (const float*)d_in[idx++];
  const float* bnw  = (const float*)d_in[idx++];
  const float* bnb  = (const float*)d_in[idx++];
  const float* dnw  = (const float*)d_in[idx++];
  const float* dnb  = (const float*)d_in[idx++];
  const float* taup = (const float*)d_in[idx++];

  float* W    = (float*)d_ws;
  float* z    = W;                         // TOKS*2
  float* xin  = z    + (size_t)TOKS * 2;   // TOKS*4
  float* buf1 = xin  + (size_t)TOKS * 4;   // TOKS*8
  float* buf2 = buf1 + (size_t)TOKS * 8;   // TOKS*40
  float* u0   = buf2 + (size_t)TOKS * 40;  // TOKS*4
  float* sA   = u0   + (size_t)TOKS * 4;   // TOKS*4 (aC)
  float* sB   = sA   + (size_t)TOKS * 4;   // TOKS*4 (bC raw)
  float* sD   = sB   + (size_t)TOKS * 4;   // TOKS*4 (dC raw)
  float* souts[3] = { sA, sB, sD };
  float* dout = (float*)d_out;

  k_prep<<<(TOKS + 255) / 256, 256, 0, stream>>>(x, z, xin);

  for (int s = 0; s < 3; ++s) {
    const float* uin = xin;
    for (int L2 = 0; L2 < 2; ++L2) {
      float* uout = (L2 == 0) ? u0 : souts[s];
      Layer lp = layers[s * 2 + L2];
      k_proj_in   <<<TILES, 32, 0, stream>>>(uin, lp, buf1);
      k_conv_xproj<<<TILES, 32, 0, stream>>>(buf1, lp, buf2);
      k_scan      <<<BSZ,   32, 0, stream>>>(uin, buf1, buf2, lp, uout);
      uin = uout;
    }
  }

  k_ctrl<<<(TOKS * 4 + 255) / 256, 256, 0, stream>>>(sA, sB, sD, dout);
  k_head<<<(TOKS + 255) / 256, 256, 0, stream>>>(sA, z, dtp, anw, anb, bnw, bnb,
                                                 dnw, dnb, taup, dout);
}
